// GraphAttentionLayer_52561809768878
// MI455X (gfx1250) — matrix operand order repaired and checked by reading, not compiled, not run on hardware
//
#include <hip/hip_runtime.h>
#include <cstdint>

#define N_NODES 4096
#define IN_FEAT 128
#define OUT_FEAT 64
#define NHEAD 4
#define HF 256  /* NHEAD * OUT_FEAT */
#define NEG_SLOPE 0.2f

typedef __attribute__((ext_vector_type(2)))  float  v2f;
typedef __attribute__((ext_vector_type(8)))  float  v8f;
typedef __attribute__((ext_vector_type(16))) __bf16 v16bf;

union B16cvt {
    struct { uint4 lo, hi; } q;
    v16bf v;
};

// ---------------------------------------------------------------------------
// K1: h = x @ W  (f32 WMMA 16x16x4), + attn_src/attn_tgt.
// Stores bf16 h TRANSPOSED: hbf_t[f_global][j]  (f-major, K-contiguous for K3).
// 256 blocks (16-row tiles) x 128 threads (wave w == head w, 64 output cols).
// ---------------------------------------------------------------------------
__global__ __launch_bounds__(128) void gat_gemm_kernel(
    const float* __restrict__ x, const float* __restrict__ W,
    const float* __restrict__ a, __bf16* __restrict__ hbf_t,
    float* __restrict__ src, float* __restrict__ tgt)
{
    __shared__ float xt[16 * IN_FEAT];   // 8 KB
    __shared__ float ht[16 * HF];        // 16 KB

    const int row0 = blockIdx.x * 16;
    const int tid  = threadIdx.x;
    const int lane = tid & 31;
    const int wv   = tid >> 5;          // head index
    const int half = lane >> 4;         // 0 or 1
    const int m    = lane & 15;

    // Stage x tile (16 x 128) coalesced.
    for (int i = tid; i < 16 * IN_FEAT / 4; i += 128)
        ((float4*)xt)[i] = ((const float4*)(x + (size_t)row0 * IN_FEAT))[i];
    __syncthreads();

    v8f C[4] = {};
    for (int k0 = 0; k0 < IN_FEAT; k0 += 4) {
        // A (16x4 f32): lane m holds K = k0 + 2*half + {0,1}.
        v2f A;
        A[0] = xt[m * IN_FEAT + k0 + 2 * half + 0];
        A[1] = xt[m * IN_FEAT + k0 + 2 * half + 1];
#pragma unroll
        for (int c = 0; c < 4; ++c) {
            const int col = wv * OUT_FEAT + c * 16 + m;
            // B (4x16 f32): element(k,n): vgpr=k&1, lane half selects k>=2.
            v2f B;
            B[0] = W[(size_t)(k0 + 2 * half + 0) * HF + col];
            B[1] = W[(size_t)(k0 + 2 * half + 1) * HF + col];
            C[c] = __builtin_amdgcn_wmma_f32_16x16x4_f32(
                false, A, false, B, (short)0, C[c], false, false);
        }
    }

    // C layout: VGPR r, lane -> row = r + 8*half, col = 16c + m.
#pragma unroll
    for (int c = 0; c < 4; ++c) {
        const int col = wv * OUT_FEAT + c * 16 + m;
#pragma unroll
        for (int r = 0; r < 8; ++r) {
            const int row = r + 8 * half;
            const float v = C[c][r];
            ht[row * HF + col] = v;
            hbf_t[(size_t)col * N_NODES + row0 + row] = (__bf16)v;  // transposed
        }
    }
    __syncthreads();

    // attn_src / attn_tgt: 64 threads, (row, head) each dot 64 features.
    if (tid < 64) {
        const int row = tid & 15, hd = tid >> 4;
        float ss = 0.f, ts = 0.f;
        for (int f = 0; f < OUT_FEAT; ++f) {
            const float v = ht[row * HF + hd * OUT_FEAT + f];
            ss += v * a[hd * 2 * OUT_FEAT + f];
            ts += v * a[hd * 2 * OUT_FEAT + OUT_FEAT + f];
        }
        src[(size_t)(row0 + row) * NHEAD + hd] = ss;
        tgt[(size_t)(row0 + row) * NHEAD + hd] = ts;
    }
}

// ---------------------------------------------------------------------------
// K2: rowmax[i][h] = max over masked j (adj[i][j]!=0 or j==i) of tgt[j][h].
// (leaky-relu is monotone, so softmax row-max = lrelu(src_i + rowmax_i))
// ---------------------------------------------------------------------------
__global__ __launch_bounds__(256) void gat_rowmax_kernel(
    const float* __restrict__ adj, const float* __restrict__ tgt,
    float* __restrict__ rowmax)
{
    __shared__ float4 red[256];
    const int i   = blockIdx.x;
    const int tid = threadIdx.x;
    float4 mv = make_float4(-3.0e38f, -3.0e38f, -3.0e38f, -3.0e38f);
    for (int j = tid; j < N_NODES; j += 256) {
        const float av = adj[(size_t)i * N_NODES + j];
        if (av != 0.f || j == i) {
            const float4 t = ((const float4*)tgt)[j];
            mv.x = fmaxf(mv.x, t.x); mv.y = fmaxf(mv.y, t.y);
            mv.z = fmaxf(mv.z, t.z); mv.w = fmaxf(mv.w, t.w);
        }
    }
    red[tid] = mv;
    __syncthreads();
    for (int s = 128; s > 0; s >>= 1) {
        if (tid < s) {
            float4 v = red[tid], o = red[tid + s];
            v.x = fmaxf(v.x, o.x); v.y = fmaxf(v.y, o.y);
            v.z = fmaxf(v.z, o.z); v.w = fmaxf(v.w, o.w);
            red[tid] = v;
        }
        __syncthreads();
    }
    if (tid == 0) ((float4*)rowmax)[i] = red[0];
}

// ---------------------------------------------------------------------------
// K3: fused exp/softmax-denominator + aggregation, bf16 WMMA 16x16x32.
// Grid (64, NHEAD): block = 64 rows of one head, 4 waves x 16-row tiles.
// Double-buffered staging: adj via async DMA-to-LDS, h tile via coalesced
// copies from the transposed bf16 h; B fragments are 2x ds_load_b128.
// ---------------------------------------------------------------------------
__global__ __launch_bounds__(128) void gat_aggregate_kernel(
    const float* __restrict__ adj, const __bf16* __restrict__ hbf_t,
    const float* __restrict__ src, const float* __restrict__ tgt,
    const float* __restrict__ rowmax, float* __restrict__ out)
{
    __shared__ float  __align__(16) sadj[2][64 * 32];       // 2 x 8 KB
    __shared__ __bf16 __align__(16) sht[2][OUT_FEAT * 32];  // 2 x 4 KB (f-major)
    __shared__ float  stgt[2][32];
    __shared__ float  ssrc[64];
    __shared__ float  smax[64];
    __shared__ float  sZ[64];

    const int head  = blockIdx.y;
    const int rb0   = blockIdx.x * 64;
    const int tid   = threadIdx.x;
    const int lane  = tid & 31;
    const int wv    = tid >> 5;
    const int half  = lane >> 4;
    const int m     = lane & 15;
    const int wrow0 = rb0 + wv * 16;     // this wave's 16-row tile

    if (tid < 64) {
        ssrc[tid] = src[(size_t)(rb0 + tid) * NHEAD + head];
        smax[tid] = rowmax[(size_t)(rb0 + tid) * NHEAD + head];
    }

    auto stage = [&](int jt, int b) {
        const int j0s = jt * 32;
        // adj tile 64x32 f32 = 512 x b128, async DMA into LDS.
#pragma unroll
        for (int it = 0; it < 4; ++it) {
            const int idx = tid + it * 128;          // 0..511
            const int r = idx >> 3, q = idx & 7;     // row, 4-float quad
            uint32_t lds = (uint32_t)(uintptr_t)&sadj[b][r * 32 + q * 4];
            uint64_t ga  = (uint64_t)(uintptr_t)(adj + (size_t)(rb0 + r) * N_NODES + j0s + q * 4);
            asm volatile("global_load_async_to_lds_b128 %0, %1, off"
                         :: "v"(lds), "v"(ga) : "memory");
        }
        // h tile, f-major: 64 f-rows x 32 j = 256 x uint4, coalesced.
        for (int i = tid; i < 256; i += 128) {
            const int f = i >> 2, q = i & 3;
            ((uint4*)&sht[b][0])[i] =
                *(const uint4*)(hbf_t + (size_t)(head * OUT_FEAT + f) * N_NODES + j0s + q * 8);
        }
        if (tid < 32) stgt[b][tid] = tgt[(size_t)(j0s + tid) * NHEAD + head];
    };

    v8f  C[4] = {};
    float zacc = 0.f;
    constexpr int NT = N_NODES / 32;

    stage(0, 0);

    for (int kt = 0; kt < NT; ++kt) {
        const int cur = kt & 1;
        const int j0  = kt * 32;

        asm volatile("s_wait_asynccnt 0x0" ::: "memory");
        __syncthreads();                       // stage(kt) visible to all waves

        if (kt + 1 < NT) stage(kt + 1, cur ^ 1);  // prefetch next tile

        // --- build A = P (16x32 bf16). A[i]: i<8 -> k = i + 8*half,
        //     i>=8 -> k = 16 + (i-8) + 8*half (ISA 16-bit A layout). ---
        const float* adjb = &sadj[cur][0];
        v16bf A;
        const float sv = ssrc[wv * 16 + m];
        const float mx = smax[wv * 16 + m];
#pragma unroll
        for (int i = 0; i < 16; ++i) {
            const int j = (i < 8) ? (i + 8 * half) : (i + 8 + 8 * half);
            const float av = adjb[(wv * 16 + m) * 32 + j];
            const bool msk = (av != 0.f) || ((wrow0 + m) == (j0 + j));
            float e = sv + stgt[cur][j];
            e = (e > 0.f) ? e : NEG_SLOPE * e;
            const float p = msk ? __expf(e - mx) : 0.f;
            zacc += p;
            A[i] = (__bf16)p;
        }

        // --- C += A @ B; B lane n=m needs elements [8*half, 8*half+8) and
        //     [16+8*half, 24+8*half) of f-row (c*16+m) -> 2x b128. ---
        const uint4* shq = (const uint4*)&sht[cur][0];
#pragma unroll
        for (int c = 0; c < 4; ++c) {
            B16cvt u;
            const int base = (c * 16 + m) * 4 + half;   // uint4 index
            u.q.lo = shq[base + 0];
            u.q.hi = shq[base + 2];
            C[c] = __builtin_amdgcn_wmma_f32_16x16x32_bf16(
                false, A, false, u.v, (short)0, C[c], false, false);
        }
    }

    // Row sums: lanes m and m^16 cover disjoint j-sets of row m.
    const float ztot = zacc + __shfl_xor(zacc, 16, 32);
    __syncthreads();
    if (lane < 16) sZ[wv * 16 + m] = ztot;
    __syncthreads();

#pragma unroll
    for (int c = 0; c < 4; ++c) {
#pragma unroll
        for (int r = 0; r < 8; ++r) {
            const int row = r + 8 * half;
            const float z = sZ[wv * 16 + row];
            const float v = (z > 0.f) ? C[c][r] / z : 0.f;
            out[(size_t)(wrow0 + row) * HF + head * OUT_FEAT + c * 16 + m] = v;
        }
    }
}

// ---------------------------------------------------------------------------
extern "C" void kernel_launch(void* const* d_in, const int* in_sizes, int n_in,
                              void* d_out, int out_size, void* d_ws, size_t ws_size,
                              hipStream_t stream)
{
    const float* x   = (const float*)d_in[0];
    const float* adj = (const float*)d_in[1];
    const float* W   = (const float*)d_in[2];
    const float* a   = (const float*)d_in[3];
    float* out = (float*)d_out;

    // Workspace: hbf_t (bf16, 2 MB, feature-major) | src | tgt | rowmax
    __bf16* hbf_t = (__bf16*)d_ws;
    float*  src = (float*)((char*)d_ws + (size_t)N_NODES * HF * sizeof(__bf16));
    float*  tgt = src + (size_t)N_NODES * NHEAD;
    float*  rmx = tgt + (size_t)N_NODES * NHEAD;

    gat_gemm_kernel<<<N_NODES / 16, 128, 0, stream>>>(x, W, a, hbf_t, src, tgt);
    gat_rowmax_kernel<<<N_NODES, 256, 0, stream>>>(adj, tgt, rmx);
    dim3 g3(N_NODES / 64, NHEAD);
    gat_aggregate_kernel<<<g3, 128, 0, stream>>>(adj, hbf_t, src, tgt, rmx, out);
}
